// Decoder_65000035058115
// MI455X (gfx1250) — matrix operand order repaired and checked by reading, not compiled, not run on hardware
//
#include <hip/hip_runtime.h>
#include <hip/hip_bf16.h>
#include <math.h>

typedef __attribute__((ext_vector_type(16))) _Float16 v16h;
typedef __attribute__((ext_vector_type(8)))  float    v8f;

#define VOCAB 50257
#define HDIM  1024
#define LSEQ  256

// ---- workspace layout (floats) ----
#define WS_X     0        // 2048
#define WS_G     2048     // 4096
#define WS_H1    6144
#define WS_C1    7168
#define WS_H2    8192
#define WS_C2    9216
#define WS_TMPP  10240
#define WS_Q     11264
#define WS_SCORE 12288    // 256
#define WS_CTX   12544
#define WS_HTT   13568
#define WS_RED1  14592    // 64
#define WS_RED2  14656    // 64
#define WS_LSE   14720
#define WS_PT    14724

// ---- output layout (floats) ----
#define OUT_H   (VOCAB)
#define OUT_C   (VOCAB + 2048)
#define OUT_HTT (VOCAB + 4096)
#define OUT_AT  (VOCAB + 5120)
#define OUT_PT  (VOCAB + 5376)

__device__ __forceinline__ float sigmoidf_(float x) { return 1.0f / (1.0f + expf(-x)); }

// x = [emb[tok], h_t_tilde]
__global__ void k_build_x(const float* __restrict__ emb, const int* __restrict__ tok,
                          const float* __restrict__ htt, float* __restrict__ x) {
    int i = blockIdx.x * blockDim.x + threadIdx.x;
    if (i >= 2048) return;
    if (i < HDIM) x[i] = emb[(size_t)tok[0] * HDIM + i];
    else          x[i] = htt[i - HDIM];
}

// g[r] = Wih[r,:]·x + bih[r] + Whh[r,:]·h + bhh[r]   (wave-per-row, float4 streams)
__global__ void k_gates(const float* __restrict__ Wih, int Kx, const float* __restrict__ x,
                        const float* __restrict__ Whh, const float* __restrict__ h,
                        const float* __restrict__ bih, const float* __restrict__ bhh,
                        float* __restrict__ g, int rows) {
    int lane = threadIdx.x & 31, wave = threadIdx.x >> 5;
    int r = blockIdx.x * (blockDim.x >> 5) + wave;
    if (r >= rows) return;
    const float* wr = Wih + (size_t)r * Kx;
    float s = 0.f;
    for (int k = lane * 4; k < Kx; k += 128) {
        float4 w = *(const float4*)(wr + k), v = *(const float4*)(x + k);
        s += w.x * v.x + w.y * v.y + w.z * v.z + w.w * v.w;
    }
    const float* wr2 = Whh + (size_t)r * HDIM;
    for (int k = lane * 4; k < HDIM; k += 128) {
        float4 w = *(const float4*)(wr2 + k), v = *(const float4*)(h + k);
        s += w.x * v.x + w.y * v.y + w.z * v.z + w.w * v.w;
    }
    for (int o = 16; o; o >>= 1) s += __shfl_down(s, o);
    if (!lane) g[r] = s + bih[r] + bhh[r];
}

// torch gate order i,f,g,o
__global__ void k_cell(const float* __restrict__ g, const float* __restrict__ c_prev,
                       float* __restrict__ h_ws, float* __restrict__ c_ws,
                       float* __restrict__ h_out, float* __restrict__ c_out) {
    int j = blockIdx.x * blockDim.x + threadIdx.x;
    if (j >= HDIM) return;
    float i = g[j], f = g[HDIM + j], gg = g[2 * HDIM + j], o = g[3 * HDIM + j];
    float c2 = sigmoidf_(f) * c_prev[j] + sigmoidf_(i) * tanhf(gg);
    float h2 = sigmoidf_(o) * tanhf(c2);
    h_ws[j] = h2; c_ws[j] = c2; h_out[j] = h2; c_out[j] = c2;
}

// out[r] = act(W[r, 0:K1]·xa + W[r, K1:K1+K2]·xb + bias[r]) ; act 1 = tanh
__global__ void k_gemv2(const float* __restrict__ W, const float* __restrict__ xa, int K1,
                        const float* __restrict__ xb, int K2, const float* __restrict__ bias,
                        float* __restrict__ out, int rows, int act) {
    int lane = threadIdx.x & 31, wave = threadIdx.x >> 5;
    int r = blockIdx.x * (blockDim.x >> 5) + wave;
    if (r >= rows) return;
    const float* wr = W + (size_t)r * (K1 + K2);
    float s = 0.f;
    for (int k = lane * 4; k < K1; k += 128) {
        float4 w = *(const float4*)(wr + k), v = *(const float4*)(xa + k);
        s += w.x * v.x + w.y * v.y + w.z * v.z + w.w * v.w;
    }
    for (int k = lane * 4; k < K2; k += 128) {
        float4 w = *(const float4*)(wr + K1 + k), v = *(const float4*)(xb + k);
        s += w.x * v.x + w.y * v.y + w.z * v.z + w.w * v.w;
    }
    for (int o = 16; o; o >>= 1) s += __shfl_down(s, o);
    if (!lane) {
        if (bias) s += bias[r];
        if (act == 1) s = tanhf(s);
        out[r] = s;
    }
}

// p_t = L * sigmoid(tmpp · wdot)
__global__ void k_pt(const float* __restrict__ tmpp, const float* __restrict__ wdot,
                     float* __restrict__ pt_ws, float* __restrict__ pt_out) {
    __shared__ float red[256];
    int t = threadIdx.x;
    float s = 0.f;
    for (int j = t; j < HDIM; j += 256) s += tmpp[j] * wdot[j];
    red[t] = s; __syncthreads();
    for (int o = 128; o; o >>= 1) { if (t < o) red[t] += red[t + o]; __syncthreads(); }
    if (!t) { float p = (float)LSEQ * sigmoidf_(red[0]); pt_ws[0] = p; pt_out[0] = p; }
}

// q[k] = sum_h h2[h] * Wbil[h,k]   (column reduction, coalesced over k)
__global__ void k_bilq(const float* __restrict__ Wbil, const float* __restrict__ h2,
                       float* __restrict__ q) {
    int k = blockIdx.x * blockDim.x + threadIdx.x;
    if (k >= HDIM) return;
    float s = 0.f;
    for (int h = 0; h < HDIM; ++h) s += h2[h] * Wbil[(size_t)h * HDIM + k];
    q[k] = s;
}

// score[l] = q · h_s[l,:]  inside window, else -FLT_MAX (finfo.min)
__global__ void k_score(const float* __restrict__ q, const float* __restrict__ h_s,
                        const float* __restrict__ ptp, float* __restrict__ score) {
    int lane = threadIdx.x & 31, wave = threadIdx.x >> 5;
    int l = blockIdx.x * (blockDim.x >> 5) + wave;
    if (l >= LSEQ) return;
    float pt = ptp[0];
    int s = (int)rintf(pt);                 // RTNE matches jnp.round
    int lo = max(s - 10, 0), hi = min(s + 10, LSEQ - 1);
    if (l < lo || l > hi) { if (!lane) score[l] = -3.4028234663852886e38f; return; }
    const float* hr = h_s + (size_t)l * HDIM;
    float acc = 0.f;
    for (int k = lane * 4; k < HDIM; k += 128) {
        float4 w = *(const float4*)(hr + k), v = *(const float4*)(q + k);
        acc += w.x * v.x + w.y * v.y + w.z * v.z + w.w * v.w;
    }
    for (int o = 16; o; o >>= 1) acc += __shfl_down(acc, o);
    if (!lane) score[l] = acc;
}

// softmax over 256, a_t = a*exp(+d^2/25) (replicates reference), context = mean over window
__global__ void k_attn(const float* __restrict__ score, const float* __restrict__ ptp,
                       const float* __restrict__ h_s, float* __restrict__ at_out,
                       float* __restrict__ ctx) {
    __shared__ float red[256];
    __shared__ float atl[256];
    int t = threadIdx.x;
    float pt = ptp[0];
    int s = (int)rintf(pt);
    int lo = max(s - 10, 0), hi = min(s + 10, LSEQ - 1);
    float win = (float)(hi - lo + 1);
    float sv = score[t];
    red[t] = sv; __syncthreads();
    for (int o = 128; o; o >>= 1) { if (t < o) red[t] = fmaxf(red[t], red[t + o]); __syncthreads(); }
    float m = red[0]; __syncthreads();
    float e = expf(sv - m);                 // masked entries underflow to exactly 0
    red[t] = e; __syncthreads();
    for (int o = 128; o; o >>= 1) { if (t < o) red[t] += red[t + o]; __syncthreads(); }
    float sum = red[0]; __syncthreads();
    bool inwin = (t >= lo) && (t <= hi);
    float d = (float)t - pt;
    float expo = inwin ? (d * d) * 0.04f : 0.f;   // (D/2)^2 = 25
    float at = (e / sum) * expf(expo);
    at_out[t] = at;
    atl[t] = at;
    __syncthreads();
    for (int h = t; h < HDIM; h += 256) {
        float acc = 0.f;
        for (int l = 0; l < LSEQ; ++l) acc += atl[l] * h_s[(size_t)l * HDIM + h];
        ctx[h] = acc / win;
    }
}

// ---- dominant op: logits = Wout @ htt + bout via V_WMMA_F32_16X16X32_F16 ----
// One wave per 16-row tile. A = fp32 weight tile converted to f16 with k = 16*(lane/16) + s
// (contiguous k per lane half, as in B); B = broadcast activation chunk (column-replicated) so
// D[:,0] = W_tile · x. fp32 accumulation in C.
__global__ void k_wout_wmma(const float* __restrict__ Wout, const float* __restrict__ htt,
                            const float* __restrict__ bout, float* __restrict__ y) {
    __shared__ __align__(32) _Float16 xh[HDIM];
    int t = threadIdx.x;
    for (int i = 0; i < 4; ++i) xh[t * 4 + i] = (_Float16)htt[t * 4 + i];
    __syncthreads();

    int lane = t & 31, wave = t >> 5;
    int tile = blockIdx.x * 8 + wave;
    int ntiles = (VOCAB + 15) / 16;
    if (tile >= ntiles) return;

    int row = tile * 16 + (lane & 15);
    int rc = min(row, VOCAB - 1);
    const float* wrow = Wout + (size_t)rc * HDIM;
    int ka = (lane < 16) ? 0 : 16;  // elements 0..7 cover K = ka..ka+7
    int kb = ka + 8;                // elements 8..15 cover K = kb..kb+7
    int xo = (lane < 16) ? 0 : 16;  // B: lo lanes hold x[k0..k0+15], hi lanes x[k0+16..31]

    v8f acc = {};
    for (int k0 = 0; k0 < HDIM; k0 += 32) {
        __builtin_prefetch(wrow + k0 + 64 + ka, 0, 0);   // stream ahead (global_prefetch_b8)
        float4 w0 = *(const float4*)(wrow + k0 + ka);
        float4 w1 = *(const float4*)(wrow + k0 + ka + 4);
        float4 w2 = *(const float4*)(wrow + k0 + kb);
        float4 w3 = *(const float4*)(wrow + k0 + kb + 4);
        v16h a;
        a[0]  = (_Float16)w0.x; a[1]  = (_Float16)w0.y; a[2]  = (_Float16)w0.z; a[3]  = (_Float16)w0.w;
        a[4]  = (_Float16)w1.x; a[5]  = (_Float16)w1.y; a[6]  = (_Float16)w1.z; a[7]  = (_Float16)w1.w;
        a[8]  = (_Float16)w2.x; a[9]  = (_Float16)w2.y; a[10] = (_Float16)w2.z; a[11] = (_Float16)w2.w;
        a[12] = (_Float16)w3.x; a[13] = (_Float16)w3.y; a[14] = (_Float16)w3.z; a[15] = (_Float16)w3.w;
        v16h b = *(const v16h*)(xh + k0 + xo);
        acc = __builtin_amdgcn_wmma_f32_16x16x32_f16(false, a, false, b,
                                                     (short)0, acc, false, false);
    }
    // Column N=0: lane 0 holds D[0..7,0] in acc[0..7], lane 16 holds D[8..15,0]
    if (lane == 0 || lane == 16) {
        int rbase = tile * 16 + (lane == 0 ? 0 : 8);
        for (int i = 0; i < 8; ++i) {
            int r = rbase + i;
            if (r < VOCAB) y[r] = acc[i] + bout[r];
        }
    }
}

// ---- log_softmax over V in 4 passes ----
__global__ void k_redmax(const float* __restrict__ y, float* __restrict__ part) {
    __shared__ float red[256];
    int t = threadIdx.x;
    float m = -3.4028234663852886e38f;
    for (int i = blockIdx.x * 256 + t; i < VOCAB; i += 64 * 256) m = fmaxf(m, y[i]);
    red[t] = m; __syncthreads();
    for (int o = 128; o; o >>= 1) { if (t < o) red[t] = fmaxf(red[t], red[t + o]); __syncthreads(); }
    if (!t) part[blockIdx.x] = red[0];
}
__global__ void k_redsum(const float* __restrict__ y, const float* __restrict__ pmax,
                         float* __restrict__ psum) {
    __shared__ float red[256];
    int t = threadIdx.x;
    float m = pmax[0];
    for (int i = 1; i < 64; ++i) m = fmaxf(m, pmax[i]);
    float s = 0.f;
    for (int i = blockIdx.x * 256 + t; i < VOCAB; i += 64 * 256) s += expf(y[i] - m);
    red[t] = s; __syncthreads();
    for (int o = 128; o; o >>= 1) { if (t < o) red[t] += red[t + o]; __syncthreads(); }
    if (!t) psum[blockIdx.x] = red[0];
}
__global__ void k_redfin(const float* __restrict__ pmax, const float* __restrict__ psum,
                         float* __restrict__ lse) {
    __shared__ float red[64];
    int t = threadIdx.x;
    float m = pmax[0];
    for (int i = 1; i < 64; ++i) m = fmaxf(m, pmax[i]);
    red[t] = psum[t]; __syncthreads();
    for (int o = 32; o; o >>= 1) { if (t < o) red[t] += red[t + o]; __syncthreads(); }
    if (!t) lse[0] = m + logf(red[0]);
}
__global__ void k_yfinal(float* __restrict__ y, const float* __restrict__ lse) {
    int i = blockIdx.x * blockDim.x + threadIdx.x;
    if (i < VOCAB) y[i] -= lse[0];
}

extern "C" void kernel_launch(void* const* d_in, const int* in_sizes, int n_in,
                              void* d_out, int out_size, void* d_ws, size_t ws_size,
                              hipStream_t stream) {
    (void)in_sizes; (void)n_in; (void)out_size; (void)ws_size;
    const int*   tok   = (const int*)  d_in[0];
    const float* h0    = (const float*)d_in[1];
    const float* c0    = (const float*)d_in[2];
    const float* h_s   = (const float*)d_in[3];
    const float* htt0  = (const float*)d_in[4];
    const float* emb   = (const float*)d_in[5];
    const float* Wp    = (const float*)d_in[6];
    const float* wdot  = (const float*)d_in[7];
    const float* Wbil  = (const float*)d_in[8];
    const float* Wcomb = (const float*)d_in[9];
    const float* bcomb = (const float*)d_in[10];
    const float* Wih0  = (const float*)d_in[11];
    const float* Whh0  = (const float*)d_in[12];
    const float* bih0  = (const float*)d_in[13];
    const float* bhh0  = (const float*)d_in[14];
    const float* Wih1  = (const float*)d_in[15];
    const float* Whh1  = (const float*)d_in[16];
    const float* bih1  = (const float*)d_in[17];
    const float* bhh1  = (const float*)d_in[18];
    const float* Wout  = (const float*)d_in[19];
    const float* bout  = (const float*)d_in[20];

    float* out = (float*)d_out;
    float* ws  = (float*)d_ws;

    k_build_x<<<8, 256, 0, stream>>>(emb, tok, htt0, ws + WS_X);
    k_gates<<<512, 256, 0, stream>>>(Wih0, 2048, ws + WS_X, Whh0, h0, bih0, bhh0, ws + WS_G, 4096);
    k_cell<<<4, 256, 0, stream>>>(ws + WS_G, c0, ws + WS_H1, ws + WS_C1, out + OUT_H, out + OUT_C);
    k_gates<<<512, 256, 0, stream>>>(Wih1, 1024, ws + WS_H1, Whh1, h0 + HDIM, bih1, bhh1, ws + WS_G, 4096);
    k_cell<<<4, 256, 0, stream>>>(ws + WS_G, c0 + HDIM, ws + WS_H2, ws + WS_C2,
                                  out + OUT_H + HDIM, out + OUT_C + HDIM);
    k_gemv2<<<128, 256, 0, stream>>>(Wp, ws + WS_H2, 1024, nullptr, 0, nullptr,
                                     ws + WS_TMPP, 1024, 1);
    k_pt<<<1, 256, 0, stream>>>(ws + WS_TMPP, wdot, ws + WS_PT, out + OUT_PT);
    k_bilq<<<4, 256, 0, stream>>>(Wbil, ws + WS_H2, ws + WS_Q);
    k_score<<<32, 256, 0, stream>>>(ws + WS_Q, h_s, ws + WS_PT, ws + WS_SCORE);
    k_attn<<<1, 256, 0, stream>>>(ws + WS_SCORE, ws + WS_PT, h_s, out + OUT_AT, ws + WS_CTX);
    k_gemv2<<<128, 256, 0, stream>>>(Wcomb, ws + WS_CTX, 1024, ws + WS_H2, 1024, bcomb,
                                     ws + WS_HTT, 1024, 1);
    hipMemcpyAsync(out + OUT_HTT, ws + WS_HTT, HDIM * sizeof(float),
                   hipMemcpyDeviceToDevice, stream);
    k_wout_wmma<<<393, 256, 0, stream>>>(Wout, ws + WS_HTT, bout, out);
    k_redmax<<<64, 256, 0, stream>>>(out, ws + WS_RED1);
    k_redsum<<<64, 256, 0, stream>>>(out, ws + WS_RED1, ws + WS_RED2);
    k_redfin<<<1, 64, 0, stream>>>(ws + WS_RED1, ws + WS_RED2, ws + WS_LSE);
    k_yfinal<<<197, 256, 0, stream>>>(out, ws + WS_LSE);
}
